// GIN_Graph_Sequence_33088428049206
// MI455X (gfx1250) — matrix operand order repaired and checked by reading, not compiled, not run on hardware
//
#include <hip/hip_runtime.h>
#include <hip/hip_bf16.h>

// ---- problem constants (from reference) ----
#define NN 100000
#define EE 1600000
#define FIN 64
#define HH 32
#define CC 2
#define GG 500
#define BN_EPS 1e-5f

typedef __attribute__((ext_vector_type(16))) _Float16 v16h;
typedef __attribute__((ext_vector_type(8)))  _Float16 h8;
typedef __attribute__((ext_vector_type(8)))  float    v8f;

union AOp { v16h v; h8 h[2]; };

// ------------------------------------------------------------------
// utility kernels
// ------------------------------------------------------------------
__global__ __launch_bounds__(256) void copyf_kernel(const float* __restrict__ s,
                                                    float* __restrict__ d, int n) {
  int i = blockIdx.x * 256 + threadIdx.x;
  if (i < n) d[i] = s[i];
}

__global__ __launch_bounds__(256) void zerof_kernel(float* __restrict__ d, int n) {
  int i = blockIdx.x * 256 + threadIdx.x;
  if (i < n) d[i] = 0.f;
}

// ------------------------------------------------------------------
// edge scatter: agg[dst] += h[src]  (F features, F/4 threads per edge)
// ------------------------------------------------------------------
template <int F>
__global__ __launch_bounds__(256) void scatter_kernel(const float* __restrict__ h,
                                                      const int* __restrict__ ei,
                                                      float* __restrict__ agg) {
  const int TPE = F / 4;
  long t = (long)blockIdx.x * 256 + threadIdx.x;
  long e = t / TPE;
  int  f = (int)(t % TPE) * 4;
  if (e >= EE) return;
  int s = ei[e];
  int d = ei[EE + e];
  const float4 v = *(const float4*)(h + (long)s * F + f);
  float* p = agg + (long)d * F + f;
  atomicAdd(p + 0, v.x);
  atomicAdd(p + 1, v.y);
  atomicAdd(p + 2, v.z);
  atomicAdd(p + 3, v.w);
}

// ------------------------------------------------------------------
// Fused GIN MLP + ReLU + BN, WMMA f16 path.
//   out = BN(ReLU( ReLU(A@W1^T + b1) @ W2^T + b2 ))
// A: [nrows, KIN] f32, W1: [32, KIN], W2: [32, 32], out: [nrows, 32]
// Block = 128 threads (4 waves); each wave owns a 16-row tile.
// ------------------------------------------------------------------
template <int KIN>
__global__ __launch_bounds__(128) void gin_mlp_kernel(
    const float* __restrict__ A, const float* __restrict__ W1,
    const float* __restrict__ b1, const float* __restrict__ W2,
    const float* __restrict__ b2, const float* __restrict__ bn_g,
    const float* __restrict__ bn_b, const float* __restrict__ bn_m,
    const float* __restrict__ bn_v, float* __restrict__ Hout, int nrows) {
  __shared__ alignas(16) _Float16 Bs1[HH * KIN];    // W1 row-major, f16
  __shared__ alignas(16) _Float16 Bs2[HH * HH];     // W2 row-major, f16
  __shared__ alignas(16) _Float16 As[4][16 * KIN];  // per-wave A tile, row-major
  __shared__ alignas(16) _Float16 Ts[4][16 * HH];   // per-wave layer-1 output

  const int tid  = threadIdx.x;
  const int wave = tid >> 5;
  const int lane = tid & 31;
  const int col  = lane & 15;     // N-column within 16-wide tile
  const bool hi  = lane >= 16;

  // stage weights (f32 -> f16)
  for (int i = tid; i < HH * KIN; i += 128) Bs1[i] = (_Float16)W1[i];
  for (int i = tid; i < HH * HH; i += 128)  Bs2[i] = (_Float16)W2[i];

  // stage A tile (16 rows per wave), coalesced
  const int rowBase = blockIdx.x * 64 + wave * 16;
  for (int i = lane; i < 16 * KIN; i += 32) {
    int r = i / KIN, c = i % KIN;
    int gr = rowBase + r;
    float v = (gr < nrows) ? A[(long)gr * KIN + c] : 0.0f;
    As[wave][i] = (_Float16)v;
  }
  __syncthreads();

  // ---- layer 1: C[16x32] = A[16xKIN] x B[KINx32], two 16-wide n-tiles ----
  v8f acc0 = {};
  v8f acc1 = {};
  const _Float16* as = &As[wave][0];
#pragma unroll
  for (int kb = 0; kb < KIN; kb += 32) {
    // A operand: lane<16 holds K=kb..kb+15, lane>=16 holds K=kb+16..kb+31
    AOp a;
    int abase = col * KIN + kb + (hi ? 16 : 0);
    a.h[0] = *(const h8*)(as + abase);
    a.h[1] = *(const h8*)(as + abase + 8);
    // B operands: lane<16 holds K=kb..kb+15, lane>=16 holds K=kb+16..kb+31
    AOp b0, b1v;
    int bb0 = col * KIN + kb + (hi ? 16 : 0);
    int bb1 = (col + 16) * KIN + kb + (hi ? 16 : 0);
    b0.h[0]  = *(const h8*)(Bs1 + bb0);
    b0.h[1]  = *(const h8*)(Bs1 + bb0 + 8);
    b1v.h[0] = *(const h8*)(Bs1 + bb1);
    b1v.h[1] = *(const h8*)(Bs1 + bb1 + 8);
    acc0 = __builtin_amdgcn_wmma_f32_16x16x32_f16(false, a.v, false, b0.v,
                                                  (short)0, acc0, false, false);
    acc1 = __builtin_amdgcn_wmma_f32_16x16x32_f16(false, a.v, false, b1v.v,
                                                  (short)0, acc1, false, false);
  }

  // bias + ReLU, re-stage into A layout for layer 2
  {
    float bb0 = b1[col], bb1 = b1[col + 16];
#pragma unroll
    for (int r = 0; r < 8; r++) {
      int m = r + (hi ? 8 : 0);
      Ts[wave][m * HH + col]      = (_Float16)fmaxf(acc0[r] + bb0, 0.f);
      Ts[wave][m * HH + col + 16] = (_Float16)fmaxf(acc1[r] + bb1, 0.f);
    }
  }
  __syncthreads();

  // ---- layer 2: D[16x32] = T[16x32] x W2^T[32x32] ----
  v8f d0 = {};
  v8f d1 = {};
  {
    const _Float16* ts = &Ts[wave][0];
    AOp a;
    int abase = col * HH + (hi ? 16 : 0);
    a.h[0] = *(const h8*)(ts + abase);
    a.h[1] = *(const h8*)(ts + abase + 8);
    AOp b0, b1v;
    int bb0 = col * HH + (hi ? 16 : 0);
    int bb1 = (col + 16) * HH + (hi ? 16 : 0);
    b0.h[0]  = *(const h8*)(Bs2 + bb0);
    b0.h[1]  = *(const h8*)(Bs2 + bb0 + 8);
    b1v.h[0] = *(const h8*)(Bs2 + bb1);
    b1v.h[1] = *(const h8*)(Bs2 + bb1 + 8);
    d0 = __builtin_amdgcn_wmma_f32_16x16x32_f16(false, a.v, false, b0.v,
                                                (short)0, d0, false, false);
    d1 = __builtin_amdgcn_wmma_f32_16x16x32_f16(false, a.v, false, b1v.v,
                                                (short)0, d1, false, false);
  }

  // epilogue: + b2, ReLU, BatchNorm (eval), store f32
  {
    int c0 = col, c1 = col + 16;
    float bias0 = b2[c0], bias1 = b2[c1];
    float s0 = bn_g[c0] * rsqrtf(bn_v[c0] + BN_EPS);
    float s1 = bn_g[c1] * rsqrtf(bn_v[c1] + BN_EPS);
    float m0 = bn_m[c0], m1 = bn_m[c1];
    float be0 = bn_b[c0], be1 = bn_b[c1];
#pragma unroll
    for (int r = 0; r < 8; r++) {
      int m = r + (hi ? 8 : 0);
      int gr = rowBase + m;
      if (gr < nrows) {
        float v0 = fmaxf(d0[r] + bias0, 0.f);
        float v1 = fmaxf(d1[r] + bias1, 0.f);
        Hout[(long)gr * HH + c0] = (v0 - m0) * s0 + be0;
        Hout[(long)gr * HH + c1] = (v1 - m1) * s1 + be1;
      }
    }
  }
}

// ------------------------------------------------------------------
// global_add_pool: g[batch[n]] += h[n]
// ------------------------------------------------------------------
__global__ __launch_bounds__(256) void pool_kernel(const float* __restrict__ h,
                                                   const int* __restrict__ batch,
                                                   float* __restrict__ g) {
  long t = (long)blockIdx.x * 256 + threadIdx.x;
  if (t >= (long)NN * HH) return;
  int n = (int)(t >> 5);
  int f = (int)(t & 31);
  atomicAdd(&g[batch[n] * HH + f], h[t]);
}

// ------------------------------------------------------------------
// dense tail: seq = (relu(g @ fc1^T + b)) @ fc2^T + b   [G, C]
// ------------------------------------------------------------------
__global__ __launch_bounds__(256) void head_kernel(
    const float* __restrict__ g, const float* __restrict__ fc1W,
    const float* __restrict__ fc1b, const float* __restrict__ fc2W,
    const float* __restrict__ fc2b, float* __restrict__ seq) {
  int i = blockIdx.x * 256 + threadIdx.x;
  if (i >= GG) return;
  float gv[HH];
#pragma unroll
  for (int k = 0; k < HH; k++) gv[k] = g[i * HH + k];
  float hid[HH];
#pragma unroll
  for (int j = 0; j < HH; j++) {
    float a = fc1b[j];
#pragma unroll
    for (int k = 0; k < HH; k++) a += fc1W[j * HH + k] * gv[k];
    hid[j] = fmaxf(a, 0.f);
  }
#pragma unroll
  for (int c = 0; c < CC; c++) {
    float a = fc2b[c];
#pragma unroll
    for (int j = 0; j < HH; j++) a += fc2W[c * HH + j] * hid[j];
    seq[i * CC + c] = a;
  }
}

// ------------------------------------------------------------------
// Bidirectional LSTM over [G, C] + reduce + log_softmax (single block)
// 256 threads: tid>>7 = direction, tid&127 = gate row (4H = 128).
// ------------------------------------------------------------------
__device__ __forceinline__ float sigf(float x) { return 1.f / (1.f + expf(-x)); }

__global__ __launch_bounds__(256) void lstm_tail_kernel(
    const float* __restrict__ seq, const float* __restrict__ WihF,
    const float* __restrict__ WhhF, const float* __restrict__ bihF,
    const float* __restrict__ bhhF, const float* __restrict__ WihB,
    const float* __restrict__ WhhB, const float* __restrict__ bihB,
    const float* __restrict__ bhhB, const float* __restrict__ redW,
    const float* __restrict__ redb, float* __restrict__ out) {
  __shared__ float sseq[GG * CC];
  __shared__ float hbuf[2][HH];
  __shared__ float zbuf[2][4 * HH];
  __shared__ float outv[CC];

  const int tid = threadIdx.x;
  const int dir = tid >> 7;
  const int r   = tid & 127;

  const float* Wih = dir ? WihB : WihF;
  const float* Whh = dir ? WhhB : WhhF;
  const float bias = dir ? (bihB[r] + bhhB[r]) : (bihF[r] + bhhF[r]);
  const float wi0 = Wih[r * 2 + 0];
  const float wi1 = Wih[r * 2 + 1];
  float wh[HH];
#pragma unroll
  for (int j = 0; j < HH; j++) wh[j] = Whh[r * HH + j];

  for (int i = tid; i < GG * CC; i += 256) sseq[i] = seq[i];
  if (r < HH) hbuf[dir][r] = 0.f;
  float c = 0.f;
  __syncthreads();

  for (int t = 0; t < GG; t++) {
    int idx = dir ? (GG - 1 - t) : t;
    float z = bias + wi0 * sseq[idx * 2] + wi1 * sseq[idx * 2 + 1];
#pragma unroll
    for (int j = 0; j < HH; j++) z += wh[j] * hbuf[dir][j];
    zbuf[dir][r] = z;
    __syncthreads();
    if (r < HH) {
      float ig = sigf(zbuf[dir][r]);
      float fg = sigf(zbuf[dir][HH + r]);
      float gg = tanhf(zbuf[dir][2 * HH + r]);
      float og = sigf(zbuf[dir][3 * HH + r]);
      c = fg * c + ig * gg;
      hbuf[dir][r] = og * tanhf(c);
    }
    __syncthreads();
  }

  if (tid < CC) {
    float a = redb[tid];
#pragma unroll
    for (int j = 0; j < HH; j++) a += redW[tid * 2 * HH + j] * hbuf[0][j];
#pragma unroll
    for (int j = 0; j < HH; j++) a += redW[tid * 2 * HH + HH + j] * hbuf[1][j];
    outv[tid] = fmaxf(a, 0.f);
  }
  __syncthreads();
  if (tid == 0) {
    float m = fmaxf(outv[0], outv[1]);
    float lse = m + logf(expf(outv[0] - m) + expf(outv[1] - m));
    out[0] = outv[0] - lse;
    out[1] = outv[1] - lse;
  }
}

// ------------------------------------------------------------------
// launch
// ------------------------------------------------------------------
extern "C" void kernel_launch(void* const* d_in, const int* in_sizes, int n_in,
                              void* d_out, int out_size, void* d_ws, size_t ws_size,
                              hipStream_t stream) {
  (void)in_sizes; (void)n_in; (void)out_size; (void)ws_size;

  const float* x        = (const float*)d_in[0];
  const int*   ei       = (const int*)d_in[1];
  const int*   batch    = (const int*)d_in[2];
  const float* gin0_W1  = (const float*)d_in[3];
  const float* gin0_b1  = (const float*)d_in[4];
  const float* gin0_W2  = (const float*)d_in[5];
  const float* gin0_b2  = (const float*)d_in[6];
  const float* bn0_g    = (const float*)d_in[7];
  const float* bn0_b    = (const float*)d_in[8];
  const float* bn0_m    = (const float*)d_in[9];
  const float* bn0_v    = (const float*)d_in[10];
  const float* gin1_W1  = (const float*)d_in[11];
  const float* gin1_b1  = (const float*)d_in[12];
  const float* gin1_W2  = (const float*)d_in[13];
  const float* gin1_b2  = (const float*)d_in[14];
  const float* bn1_g    = (const float*)d_in[15];
  const float* bn1_b    = (const float*)d_in[16];
  const float* bn1_m    = (const float*)d_in[17];
  const float* bn1_v    = (const float*)d_in[18];
  const float* fc1_W    = (const float*)d_in[19];
  const float* fc1_b    = (const float*)d_in[20];
  const float* fc2_W    = (const float*)d_in[21];
  const float* fc2_b    = (const float*)d_in[22];
  const float* WihF     = (const float*)d_in[23];
  const float* WhhF     = (const float*)d_in[24];
  const float* bihF     = (const float*)d_in[25];
  const float* bhhF     = (const float*)d_in[26];
  const float* WihB     = (const float*)d_in[27];
  const float* WhhB     = (const float*)d_in[28];
  const float* bihB     = (const float*)d_in[29];
  const float* bhhB     = (const float*)d_in[30];
  const float* redW     = (const float*)d_in[31];
  const float* redb     = (const float*)d_in[32];

  float* ws   = (float*)d_ws;
  float* agg0 = ws;                            // N*64
  float* h0   = agg0 + (size_t)NN * FIN;       // N*32
  float* agg1 = h0 + (size_t)NN * HH;          // N*32
  float* h1   = agg1 + (size_t)NN * HH;        // N*32
  float* gbuf = h1 + (size_t)NN * HH;          // G*32
  float* seq  = gbuf + (size_t)GG * HH;        // G*2

  const int mlpBlocks = (NN + 63) / 64;

  // GIN layer 0
  copyf_kernel<<<(NN * FIN + 255) / 256, 256, 0, stream>>>(x, agg0, NN * FIN);
  {
    long t = (long)EE * (FIN / 4);
    scatter_kernel<FIN><<<(int)((t + 255) / 256), 256, 0, stream>>>(x, ei, agg0);
  }
  gin_mlp_kernel<FIN><<<mlpBlocks, 128, 0, stream>>>(
      agg0, gin0_W1, gin0_b1, gin0_W2, gin0_b2, bn0_g, bn0_b, bn0_m, bn0_v, h0, NN);

  // GIN layer 1
  copyf_kernel<<<(NN * HH + 255) / 256, 256, 0, stream>>>(h0, agg1, NN * HH);
  {
    long t = (long)EE * (HH / 4);
    scatter_kernel<HH><<<(int)((t + 255) / 256), 256, 0, stream>>>(h0, ei, agg1);
  }
  gin_mlp_kernel<HH><<<mlpBlocks, 128, 0, stream>>>(
      agg1, gin1_W1, gin1_b1, gin1_W2, gin1_b2, bn1_g, bn1_b, bn1_m, bn1_v, h1, NN);

  // pool + dense head
  zerof_kernel<<<(GG * HH + 255) / 256, 256, 0, stream>>>(gbuf, GG * HH);
  {
    long t = (long)NN * HH;
    pool_kernel<<<(int)((t + 255) / 256), 256, 0, stream>>>(h1, batch, gbuf);
  }
  head_kernel<<<(GG + 255) / 256, 256, 0, stream>>>(gbuf, fc1_W, fc1_b, fc2_W,
                                                    fc2_b, seq);

  // BiLSTM + reduce + log_softmax
  lstm_tail_kernel<<<1, 256, 0, stream>>>(seq, WihF, WhhF, bihF, bhhF, WihB,
                                          WhhB, bihB, bhhB, redW, redb,
                                          (float*)d_out);
}
